// MLPPredictor_2010044695331
// MI455X (gfx1250) — matrix operand order repaired and checked by reading, not compiled, not run on hardware
//
#include <hip/hip_runtime.h>
#include <hip/hip_bf16.h>

typedef __attribute__((ext_vector_type(16))) __bf16 v16bf;
typedef __attribute__((ext_vector_type(8)))  float  v8f;
typedef unsigned short u16;
typedef unsigned int   u32;

union AU { v16bf v; uint4 q[2]; };
union BU { v16bf v; u16 s[16]; };

// ---------------- prep: fp32 -> bf16 (RNE) for h and W1 ----------------
__global__ void prep_bf16_kernel(const float* __restrict__ h,
                                 const float* __restrict__ W1,
                                 u16* __restrict__ hbf,
                                 u16* __restrict__ w1bf,
                                 int nh, int nw) {
  int i = blockIdx.x * blockDim.x + threadIdx.x;
  int total = nh + nw;
  int stride = gridDim.x * blockDim.x;
  for (; i < total; i += stride) {
    float f;
    u16* dptr;
    int o;
    if (i < nh) { f = h[i];      dptr = hbf;  o = i; }
    else        { f = W1[i-nh];  dptr = w1bf; o = i - nh; }
    u32 u = __float_as_uint(f);
    u32 r = (u + 0x7FFFu + ((u >> 16) & 1u)) >> 16;   // round-to-nearest-even
    dptr[o] = (u16)r;
  }
}

// ---------------- main: per-edge MLP via bf16 WMMA ----------------
// One wave handles a tile of 16 edges; software-pipelined gathers.
//   A (16x128 bf16) = [h[src] | h[dst]] gathered in WMMA A-layout
//   B (128x64 bf16) = W1, held entirely in registers (16 fragments)
//   C (16x64 f32)   = hidden pre-activation; ReLU + dot(W2) done in-register.
__launch_bounds__(256)
__global__ void edge_mlp_kernel(const int* __restrict__ src,
                                const int* __restrict__ dst,
                                const u16* __restrict__ hbf,
                                const u16* __restrict__ w1bf,
                                const float* __restrict__ b1,
                                const float* __restrict__ w2,
                                const float* __restrict__ b2p,
                                float* __restrict__ out,
                                int ntiles) {
  const int lane   = threadIdx.x & 31;
  const int wave   = (blockIdx.x * blockDim.x + threadIdx.x) >> 5;
  const int nwaves = (gridDim.x * blockDim.x) >> 5;
  const int lm     = lane & 15;     // column / row-in-tile selector
  const int hh     = lane >> 4;     // 0 = lanes 0-15, 1 = lanes 16-31

  // ---- Load all of W1 into B-fragments (bf16, VGPR-resident) ----
  // B 32x16 layout: element i holds K = kb*32 + i + 8*((i>=8) + (lane>=16)) (matches A),
  // column n = nb*16 + (lane & 15), packed 2 bf16 per VGPR in K order.
  v16bf B[4][4];
#pragma unroll
  for (int kb = 0; kb < 4; ++kb) {
#pragma unroll
    for (int nb = 0; nb < 4; ++nb) {
      BU bu;
      const int n = nb * 16 + lm;
#pragma unroll
      for (int i = 0; i < 16; ++i) {
        const int K = kb * 32 + i + 8 * ((i >= 8) + hh);
        bu.s[i] = w1bf[K * 64 + n];
      }
      B[kb][nb] = bu.v;
    }
  }

  float b1v[4], w2v[4];
#pragma unroll
  for (int nb = 0; nb < 4; ++nb) {
    b1v[nb] = b1[nb * 16 + lm];
    w2v[nb] = w2[nb * 16 + lm];
  }
  const float b2v = b2p[0];

  // ---- software pipeline: indices one tile ahead, rows prefetched ----
  int tile = wave;
  int sn = 0, dn = 0;
  if (tile < ntiles) {
    sn = src[tile * 16 + lm];
    dn = dst[tile * 16 + lm];
  }

  for (; tile < ntiles; tile += nwaves) {
    const int nt = tile + nwaves;
    int sn2 = sn, dn2 = dn;
    if (nt < ntiles) {
      // next tile's indices: in flight under this tile's WMMA work
      sn2 = src[nt * 16 + lm];
      dn2 = dst[nt * 16 + lm];
    }

    const uint4* srow = (const uint4*)(hbf + (size_t)sn * 64);  // 8 x 16B chunks
    const uint4* drow = (const uint4*)(hbf + (size_t)dn * 64);

    // ---- Assemble A-fragments per CDNA5 16-bit A 16x32 layout ----
    // element i: K = kb*32 + i + 8*((i>=8) + (lane>=16))
    //  -> chunk0 = feats[f0 .. f0+7], chunk1 = feats[f0+16 .. f0+23]
    //     with f0 = (kb&1)*32 + 8*hh inside the selected node row (64 feats).
    v16bf A[4];
#pragma unroll
    for (int kb = 0; kb < 4; ++kb) {
      const uint4* row = (kb < 2) ? srow : drow;  // K<64 -> src half, else dst half
      const int c0 = (kb & 1) * 4 + hh;           // chunk index (8 bf16 per chunk)
      AU au;
      au.q[0] = row[c0];
      au.q[1] = row[c0 + 2];
      A[kb] = au.v;
    }

    // ---- Layer 1 (WMMA) fused with ReLU + Layer 2 partial dot ----
    float s[8];
#pragma unroll
    for (int j = 0; j < 8; ++j) s[j] = 0.0f;

#pragma unroll
    for (int nb = 0; nb < 4; ++nb) {
      v8f acc = {};
#pragma unroll
      for (int kb = 0; kb < 4; ++kb) {
        acc = __builtin_amdgcn_wmma_f32_16x16x32_bf16(
            /*neg_a=*/false, A[kb], /*neg_b=*/false, B[kb][nb],
            /*c_mod=*/(short)0, acc, /*reuse_a=*/false, /*reuse_b=*/false);
      }
      // acc[j] at this lane = hidden pre-act of row m = j + 8*hh, col n = nb*16+lm
#pragma unroll
      for (int j = 0; j < 8; ++j) {
        float hv = acc[j] + b1v[nb];
        hv = hv > 0.0f ? hv : 0.0f;   // ReLU
        s[j] += hv * w2v[nb];          // partial dot with W2
      }
    }

    // ---- Reduce over the 16 columns (butterfly within each 16-lane half) ----
#pragma unroll
    for (int j = 0; j < 8; ++j) {
      s[j] += __shfl_xor(s[j], 1, 32);
      s[j] += __shfl_xor(s[j], 2, 32);
      s[j] += __shfl_xor(s[j], 4, 32);
      s[j] += __shfl_xor(s[j], 8, 32);
    }

    // lanes 0-7 write rows 0-7, lanes 16-23 write rows 8-15
    if (lm < 8) {
      float val = s[0];
#pragma unroll
      for (int j = 1; j < 8; ++j) val = (lm == j) ? s[j] : val;
      out[tile * 16 + hh * 8 + lm] = val + b2v;
    }

    // ---- warm WGP$/L0 with next tile's rows (global_prefetch_b8, no LOADcnt) ----
    if (nt < ntiles) {
      const char* sp = (const char*)(hbf + (size_t)sn2 * 64);   // 128-B row
      const char* dp = (const char*)(hbf + (size_t)dn2 * 64);
      __builtin_prefetch(sp,      0, 3);
      __builtin_prefetch(sp + 64, 0, 3);
      __builtin_prefetch(dp,      0, 3);
      __builtin_prefetch(dp + 64, 0, 3);
    }
    sn = sn2;
    dn = dn2;
  }
}

extern "C" void kernel_launch(void* const* d_in, const int* in_sizes, int n_in,
                              void* d_out, int out_size, void* d_ws, size_t ws_size,
                              hipStream_t stream) {
  const float* h   = (const float*)d_in[0];
  const float* W1  = (const float*)d_in[1];
  const float* b1  = (const float*)d_in[2];
  const float* W2  = (const float*)d_in[3];
  const float* b2  = (const float*)d_in[4];
  const int*   src = (const int*)d_in[5];
  const int*   dst = (const int*)d_in[6];

  const int nh  = in_sizes[0];   // N_NODES * 64
  const int nw1 = in_sizes[1];   // 128 * 64
  const int nE  = in_sizes[5];   // N_EDGES

  u16* hbf  = (u16*)d_ws;
  u16* w1bf = (u16*)((char*)d_ws + (size_t)nh * sizeof(u16));

  // 1) convert h and W1 to bf16 in scratch (halves gather traffic, feeds WMMA)
  {
    int total  = nh + nw1;
    int blocks = (total + 255) / 256;
    if (blocks > 16384) blocks = 16384;
    prep_bf16_kernel<<<blocks, 256, 0, stream>>>(h, W1, hbf, w1bf, nh, nw1);
  }

  // 2) fused gather + MLP, one wave per 16-edge tile, persistent waves
  {
    int ntiles = nE / 16;
    edge_mlp_kernel<<<512, 256, 0, stream>>>(src, dst, hbf, w1bf,
                                             b1, W2, b2, (float*)d_out, ntiles);
  }
}
